// ContrastiveLoss_7413113553592
// MI455X (gfx1250) — matrix operand order repaired and checked by reading, not compiled, not run on hardware
//
#include <hip/hip_runtime.h>
#include <hip/hip_bf16.h>

#define B_N   8192
#define D_K   1024
#define TOPK  10
#define TEMP_INV 20.0f     // 1 / 0.05
#define DELTA    0.1f

#define CHUNK      1024
#define SIM_STRIDE (CHUNK + 4)   // +4 dwords: kills bank conflicts on C stores
#define A_STRIDE   (D_K + 16)    // bf16 elements; keeps 16B alignment, skews banks

typedef __attribute__((ext_vector_type(16))) __bf16 v16bf;
typedef __attribute__((ext_vector_type(8)))  __bf16 v8bf;
typedef __attribute__((ext_vector_type(4)))  __bf16 v4bf;
typedef __attribute__((ext_vector_type(8)))  float  v8f;

// ---------------- fp32 -> bf16 conversion of doc embeddings -----------------
__global__ void cvt_bf16_kernel(const float* __restrict__ src,
                                __bf16* __restrict__ dst, int n) {
    int i = (blockIdx.x * blockDim.x + threadIdx.x) * 4;
    if (i + 3 < n) {
        float4 f = *(const float4*)(src + i);
        v4bf o;
        o[0] = (__bf16)f.x; o[1] = (__bf16)f.y;
        o[2] = (__bf16)f.z; o[3] = (__bf16)f.w;
        *(v4bf*)(dst + i) = o;
    }
}

static __device__ __forceinline__ v16bf cat8(v8bf lo, v8bf hi) {
    v16bf r;
#pragma unroll
    for (int i = 0; i < 8; ++i) { r[i] = lo[i]; r[i + 8] = hi[i]; }
    return r;
}

// ---------------- fused sim + top-10 + per-row InfoNCE ----------------------
__global__ __launch_bounds__(256)
void infonce_main_kernel(const float* __restrict__ Q,
                         const __bf16* __restrict__ Dbf,
                         float* __restrict__ rowLoss) {
    __shared__ __align__(16) __bf16 aTile[16 * A_STRIDE];
    __shared__ __align__(16) float  simBuf[16 * SIM_STRIDE];
    __shared__ float top10[16][TOPK];
    __shared__ float diag[16];

    const int tid     = threadIdx.x;
    const int wave    = tid >> 5;
    const int lane    = tid & 31;
    const int rowBase = blockIdx.x * 16;

    // Stage A strip: 16 rows of Q, converted fp32->bf16 into LDS.
    for (int i = tid; i < 16 * D_K; i += 256) {
        int r = i >> 10, k = i & (D_K - 1);
        aTile[r * A_STRIDE + k] = (__bf16)Q[(size_t)(rowBase + r) * D_K + k];
    }
    if (tid < 16 * TOPK) top10[tid / TOPK][tid % TOPK] = -__builtin_inff();
    __syncthreads();

    const int r16 = lane & 15;      // row (A) / col (B) within a 16-tile
    const int sel = lane >> 4;      // half-wave select
    // A 16x32 bf16 layout: V0-7 = K[sel*16 .. +15] (contiguous, matches B)
    const __bf16* aB = aTile + r16 * A_STRIDE + sel * 16;
    const int rowb = 8 * sel;       // C layout: lanes16-31 hold M = v + 8

    for (int chunk = 0; chunk < B_N / CHUNK; ++chunk) {
        const int chunkBase = chunk * CHUNK;

        // ---- GEMM: 16 x CHUNK block of sim via WMMA bf16 (fp32 accum) ----
        // Each wave computes TWO adjacent 16x16 tiles sharing the A fragment:
        // 1 LDS-b128 + 2 global-b128 per WMMA, two independent accum chains.
#pragma unroll 1
        for (int tp = 0; tp < CHUNK / (16 * 16); ++tp) {   // 4 passes
            const int t0  = tp * 16 + wave * 2;            // first of 2 tiles
            const int lc0 = t0 * 16 + r16;                 // local col, tile 0
            const int lc1 = lc0 + 16;                      // local col, tile 1
            const __bf16* bB0 =
                Dbf + (size_t)(chunkBase + lc0) * D_K + sel * 16;
            const __bf16* bB1 = bB0 + (size_t)16 * D_K;
            v8f c0 = {}, c1 = {};
#pragma unroll 2
            for (int k = 0; k < D_K; k += 32) {
                v16bf a  = cat8(*(const v8bf*)(aB + k),
                                *(const v8bf*)(aB + k + 8));
                v16bf b0 = *(const v16bf*)(bB0 + k);
                v16bf b1 = *(const v16bf*)(bB1 + k);
                c0 = __builtin_amdgcn_wmma_f32_16x16x32_bf16(
                         false, a, false, b0, (short)0, c0, false, false);
                c1 = __builtin_amdgcn_wmma_f32_16x16x32_bf16(
                         false, a, false, b1, (short)0, c1, false, false);
            }
            // Unconditional C store (no divergence; diagonal patched later).
#pragma unroll
            for (int v = 0; v < 8; ++v) {
                simBuf[(rowb + v) * SIM_STRIDE + lc0] = c0[v];
                simBuf[(rowb + v) * SIM_STRIDE + lc1] = c1[v];
            }
        }
        __syncthreads();

        // ---- patch diagonal: capture positive logit, zero the slot ----
        // All 16 diagonal columns of this block live in one chunk (16 | 1024).
        if (tid < 16 && (rowBase >> 10) == chunk) {
            const int lc = rowBase + tid - chunkBase;
            diag[tid] = simBuf[tid * SIM_STRIDE + lc];
            simBuf[tid * SIM_STRIDE + lc] = 0.0f;
        }
        __syncthreads();

        // ---- exact top-10 of this chunk, 2 rows per wave ----
        for (int rr = 0; rr < 2; ++rr) {
            const int r = wave * 2 + rr;
            float* simRow = simBuf + r * SIM_STRIDE;
            float ct[TOPK];
            for (int t = 0; t < TOPK; ++t) {
                float best = -__builtin_inff();
                int   bidx = 0x7fffffff;
                for (int j = lane; j < CHUNK; j += 32) {   // bank-conflict-free
                    float v = simRow[j];
                    if (v > best) { best = v; bidx = j; }
                }
#pragma unroll
                for (int off = 16; off >= 1; off >>= 1) {
                    float ov = __shfl_xor(best, off);
                    int   oi = __shfl_xor(bidx, off);
                    if (ov > best || (ov == best && oi < bidx)) {
                        best = ov; bidx = oi;
                    }
                }
                if (lane == 0 && bidx != 0x7fffffff)
                    simRow[bidx] = -__builtin_inff();      // extract
                ct[t] = best;                              // descending
            }
            if (lane == 0) {
                // merge two descending 10-lists -> running global top-10
                float merged[TOPK];
                int i = 0, j = 0;
#pragma unroll
                for (int t = 0; t < TOPK; ++t) {
                    float a = (i < TOPK) ? top10[r][i] : -__builtin_inff();
                    float b = (j < TOPK) ? ct[j]       : -__builtin_inff();
                    if (a >= b) { merged[t] = a; ++i; }
                    else        { merged[t] = b; ++j; }
                }
#pragma unroll
                for (int t = 0; t < TOPK; ++t) top10[r][t] = merged[t];
            }
        }
        __syncthreads();
    }

    // ---- per-row InfoNCE term (matches reference, incl. exp(0)=1 slots) ----
    if (tid < 16) {
        const int r = tid;
        const float logit = diag[r];
        const float el = expf(logit * TEMP_INV);
        float denom = el;
#pragma unroll
        for (int t = 0; t < TOPK; ++t) {
            float v  = top10[r][t];
            float hn = (v > logit + DELTA) ? v : 0.0f;
            denom += expf(hn * TEMP_INV);
        }
        rowLoss[rowBase + r] = -logf(el / denom);
    }
}

// ---------------- fixed-order mean over 8192 row losses ---------------------
__global__ void reduce_kernel(const float* __restrict__ rowLoss,
                              float* __restrict__ out) {
    __shared__ float s[256];
    float acc = 0.0f;
    for (int i = threadIdx.x; i < B_N; i += 256) acc += rowLoss[i];
    s[threadIdx.x] = acc;
    __syncthreads();
    for (int off = 128; off > 0; off >>= 1) {
        if ((int)threadIdx.x < off) s[threadIdx.x] += s[threadIdx.x + off];
        __syncthreads();
    }
    if (threadIdx.x == 0) out[0] = s[0] / (float)B_N;
}

extern "C" void kernel_launch(void* const* d_in, const int* in_sizes, int n_in,
                              void* d_out, int out_size, void* d_ws, size_t ws_size,
                              hipStream_t stream) {
    const float* Q  = (const float*)d_in[0];   // query_embeds (8192,1024) fp32
    const float* Dm = (const float*)d_in[1];   // doc_embeds   (8192,1024) fp32

    // workspace layout: [0,16MB) bf16 doc matrix, then 8192 row losses
    __bf16* Dbf     = (__bf16*)d_ws;
    float*  rowLoss = (float*)((char*)d_ws + (size_t)B_N * D_K * sizeof(__bf16));

    const int nElem = B_N * D_K;
    cvt_bf16_kernel<<<(nElem / 4 + 255) / 256, 256, 0, stream>>>(Dm, Dbf, nElem);
    infonce_main_kernel<<<B_N / 16, 256, 0, stream>>>(Q, Dbf, rowLoss);
    reduce_kernel<<<1, 256, 0, stream>>>(rowLoss, (float*)d_out);
}
